// LoRATransformerClassifier_44341242364416
// MI455X (gfx1250) — compile-verified, matrix operand order checked by reading, not run on hardware
//
#include <hip/hip_runtime.h>
#include <hip/hip_bf16.h>

// ---------------------------------------------------------------------------
// Problem constants (from the reference)
// ---------------------------------------------------------------------------
constexpr int V   = 32000;
constexpr int D   = 768;
constexpr int H   = 12;
constexpr int HID = 3072;
constexpr int L   = 2;
constexpr int C   = 2;
constexpr int R   = 8;
constexpr int Bb  = 64;      // batch
constexpr int S   = 512;     // seq
constexpr int HD  = D / H;   // 64
constexpr int M   = Bb * S;  // 32768 token rows
constexpr float SCALE = 16.0f / 8.0f;   // alpha / r = 2

typedef __attribute__((ext_vector_type(16))) __bf16 v16bf;
typedef __attribute__((ext_vector_type(8)))  __bf16 v8bf;
typedef __attribute__((ext_vector_type(8)))  float  v8f;

// ---------------------------------------------------------------------------
// CDNA5 16-bit WMMA fragment load (row-major source, M x K or N x K):
//  lane 0-15  : row = row0+lane,     K = k0+[0..7]  and k0+[16..23]
//  lane 16-31 : row = row0+lane-16,  K = k0+[8..15] and k0+[24..31]
// Two 16-byte loads per fragment.
// ---------------------------------------------------------------------------
__device__ __forceinline__ v16bf frag_rowmajor(const __bf16* __restrict__ p,
                                               int ld, int row0, int k0, int lane)
{
    const int r  = row0 + (lane & 15);
    const int kb = k0 + ((lane >> 4) << 3);
    const __bf16* q = p + (size_t)r * ld + kb;
    v8bf lo = *(const v8bf*)q;            // K = kb .. kb+7
    v8bf hi = *(const v8bf*)(q + 16);     // K = kb+16 .. kb+23
    v16bf out;
#pragma unroll
    for (int i = 0; i < 8; ++i) { out[i] = lo[i]; out[i + 8] = hi[i]; }
    return out;
}

// ---------------------------------------------------------------------------
// Fold LoRA into the dense weight:  out[o,k] = W[o,k] + SCALE * sum_r B[o,r]*A[r,k]
// ---------------------------------------------------------------------------
__global__ __launch_bounds__(256)
void fold_lora_bf16(const float* __restrict__ W, const float* __restrict__ A,
                    const float* __restrict__ Bm, __bf16* __restrict__ out,
                    int dO, int dI)
{
    size_t i = (size_t)blockIdx.x * blockDim.x + threadIdx.x;
    if (i >= (size_t)dO * dI) return;
    int o = (int)(i / dI), k = (int)(i % dI);
    float v = W[i];
#pragma unroll
    for (int r = 0; r < R; ++r)
        v += SCALE * Bm[(size_t)o * R + r] * A[(size_t)r * dI + k];
    out[i] = (__bf16)v;
}

__global__ __launch_bounds__(256)
void fold_lora_f32(const float* __restrict__ W, const float* __restrict__ A,
                   const float* __restrict__ Bm, float* __restrict__ out,
                   int dO, int dI)
{
    size_t i = (size_t)blockIdx.x * blockDim.x + threadIdx.x;
    if (i >= (size_t)dO * dI) return;
    int o = (int)(i / dI), k = (int)(i % dI);
    float v = W[i];
#pragma unroll
    for (int r = 0; r < R; ++r)
        v += SCALE * Bm[(size_t)o * R + r] * A[(size_t)r * dI + k];
    out[i] = v;
}

// ---------------------------------------------------------------------------
// LoRA embedding gather: row m = s*B + b ; x[m,:] = embW[tok] + SCALE*(embB @ a)
// ---------------------------------------------------------------------------
__global__ __launch_bounds__(256)
void embed_kernel(const int* __restrict__ tokens, const float* __restrict__ embW,
                  const float* __restrict__ embA, const float* __restrict__ embB,
                  float* __restrict__ xf, __bf16* __restrict__ xb)
{
    const int m = blockIdx.x;            // s*B + b
    const int s = m / Bb, b = m % Bb;
    const int t = tokens[b * S + s];
    const int tid = threadIdx.x;
    __shared__ float a[R];
    if (tid < R) a[tid] = embA[(size_t)tid * V + t];
    __syncthreads();
#pragma unroll
    for (int i = 0; i < 3; ++i) {
        int d = tid + i * 256;
        float v = embW[(size_t)t * D + d];
#pragma unroll
        for (int r = 0; r < R; ++r)
            v += SCALE * a[r] * embB[(size_t)d * R + r];
        xf[(size_t)m * D + d] = v;
        xb[(size_t)m * D + d] = (__bf16)v;
    }
}

// ---------------------------------------------------------------------------
// WMMA GEMM:  Y(MxN) = Xbf(MxK) @ Wbf(NxK)^T + bias   [+ReLU] -> f32 / bf16
// 256 threads = 8 waves; wave tile 32x32 (2x2 WMMA accs); block tile 64x128.
// Weight tile (128 x 32 K) is staged into a double-buffered LDS tile with the
// CDNA5 async global->LDS DMA path (GLOBAL_LOAD_ASYNC_TO_LDS_B128 / ASYNCcnt),
// overlapping the next K-step's weight DMA with the current WMMAs.
// The LDS destination pointer is passed to the asm as a dummy operand so the
// compiler knows `wtile` escapes and is written by the DMA (keeps ds_loads live).
// ---------------------------------------------------------------------------
__global__ __launch_bounds__(256)
void gemm_bf16_wmma(const __bf16* __restrict__ X, const __bf16* __restrict__ W,
                    const float* __restrict__ bias,
                    float* __restrict__ Yf, __bf16* __restrict__ Yb,
                    int Mdim, int Ndim, int Kdim, int relu)
{
    // Single shared object in this kernel -> LDS offset 0.
    __shared__ __bf16 wtile[2][128 * 32];   // 8 KB per buffer, double-buffered

    const int lane = threadIdx.x & 31;
    const int wave = threadIdx.x >> 5;
    const int wr = wave >> 2;                // 0..1
    const int wc = wave & 3;                 // 0..3
    const int m0   = blockIdx.y * 64 + wr * 32;
    const int nblk = blockIdx.x * 128;
    const int nloc = wc * 32;                // wave's column base inside LDS tile

    // Issue one async 16-byte global->LDS transfer for chunk c of tile `buf`.
    auto stage_one = [&](int k0, int buf, int c) {
        const int row = c >> 2;              // 0..127
        const int kc  = (c & 3) << 3;        // 0,8,16,24
        const unsigned lds =
            (unsigned)(buf * (128 * 32 * 2) + (row * 32 + kc) * 2);
        const unsigned long long ga = (unsigned long long)(uintptr_t)
            (W + (size_t)(nblk + row) * Kdim + k0 + kc);
        const __bf16* dst = &wtile[buf][row * 32 + kc];   // escapes wtile
        asm volatile("global_load_async_to_lds_b128 %0, %1, off"
                     :: "v"(lds), "v"(ga), "v"(dst) : "memory");
    };
    // 512 chunks of 16B per 128x32 tile, 2 per thread, always in range.
    auto stage_w = [&](int k0, int buf) {
        stage_one(k0, buf, (int)threadIdx.x);
        stage_one(k0, buf, (int)threadIdx.x + 256);
    };

    v8f acc[2][2] = {{{}, {}}, {{}, {}}};

    stage_w(0, 0);
    for (int k0 = 0; k0 < Kdim; k0 += 32) {
        const int buf = (k0 >> 5) & 1;

        // Wait for this K-step's weight tile DMA, then make it block-visible.
        asm volatile("s_wait_asynccnt 0x0" ::: "memory");
        __syncthreads();

        // Kick off the DMA for the next K-step under this step's math.
        if (k0 + 32 < Kdim) {
            stage_w(k0 + 32, buf ^ 1);
            __builtin_prefetch(X + (size_t)(m0 + (lane & 15)) * Kdim + k0 + 32, 0, 3);
        }

        // A fragments straight from global (b128 loads).
        v16bf a0 = frag_rowmajor(X, Kdim, m0,      k0, lane);
        v16bf a1 = frag_rowmajor(X, Kdim, m0 + 16, k0, lane);

        // B fragments from the LDS weight tile (ds reads).
        v16bf b0, b1;
        {
            const int kb = (lane >> 4) << 3;
            const int r0 = nloc + (lane & 15);
            const int r1 = r0 + 16;
#pragma unroll
            for (int i = 0; i < 8; ++i) {
                b0[i]     = wtile[buf][r0 * 32 + kb + i];
                b0[i + 8] = wtile[buf][r0 * 32 + kb + 16 + i];
                b1[i]     = wtile[buf][r1 * 32 + kb + i];
                b1[i + 8] = wtile[buf][r1 * 32 + kb + 16 + i];
            }
        }

        acc[0][0] = __builtin_amdgcn_wmma_f32_16x16x32_bf16(false, a0, false, b0, (short)0, acc[0][0], false, false);
        acc[0][1] = __builtin_amdgcn_wmma_f32_16x16x32_bf16(false, a0, false, b1, (short)0, acc[0][1], false, false);
        acc[1][0] = __builtin_amdgcn_wmma_f32_16x16x32_bf16(false, a1, false, b0, (short)0, acc[1][0], false, false);
        acc[1][1] = __builtin_amdgcn_wmma_f32_16x16x32_bf16(false, a1, false, b1, (short)0, acc[1][1], false, false);
    }

    const int nlo = lane & 15;
    const int mhi = (lane >> 4) << 3;        // 0 or 8
#pragma unroll
    for (int ti = 0; ti < 2; ++ti) {
#pragma unroll
        for (int tj = 0; tj < 2; ++tj) {
            const int mt = m0 + ti * 16 + mhi;
            const int nt = nblk + nloc + tj * 16 + nlo;
            const float bv = bias ? bias[nt] : 0.0f;
#pragma unroll
            for (int v = 0; v < 8; ++v) {
                float val = acc[ti][tj][v] + bv;
                if (relu) val = fmaxf(val, 0.0f);
                size_t idx = (size_t)(mt + v) * Ndim + nt;
                if (Yf) Yf[idx] = val;
                if (Yb) Yb[idx] = (__bf16)val;
            }
        }
    }
}

// ---------------------------------------------------------------------------
// Attention over the batch axis (per reference): for each (s,h) a 64x64 head.
// Block = 128 threads = 4 waves; wave w owns 16 score rows (4 WMMA tiles wide).
// scores -> LDS -> softmax -> P(bf16) in LDS ; V^T staged in LDS ; PV via WMMA.
// ---------------------------------------------------------------------------
__global__ __launch_bounds__(128)
void attention_wmma(const __bf16* __restrict__ Qb, const __bf16* __restrict__ Kb,
                    const __bf16* __restrict__ Vb, __bf16* __restrict__ Ob)
{
    const int sh = blockIdx.x;               // 0 .. S*H-1
    const int s = sh / H;
    const int h = sh % H;
    const int tid = threadIdx.x;
    const int lane = tid & 31;
    const int wave = tid >> 5;               // 0..3
    const int m0 = wave * 16;

    __shared__ float  scS[64 * 68];          // padded scores
    __shared__ __bf16 Pm[64 * 64];           // softmax probs
    __shared__ __bf16 Vt[64 * 64];           // V transposed: Vt[hd][c]

    const size_t base = ((size_t)s * Bb) * D + (size_t)h * HD;

    // stage V^T into LDS
    for (int i = tid; i < 64 * 64; i += 128) {
        int c = i >> 6, hd = i & 63;
        Vt[hd * 64 + c] = Vb[base + (size_t)c * D + hd];
    }

    // scores = Q @ K^T  (M=b rows, N=c cols, K=hd)
    v8f sc[4] = {{}, {}, {}, {}};
#pragma unroll
    for (int k0 = 0; k0 < HD; k0 += 32) {
        v16bf aq = frag_rowmajor(Qb + base, D, m0, k0, lane);
#pragma unroll
        for (int j = 0; j < 4; ++j) {
            v16bf bk = frag_rowmajor(Kb + base, D, j * 16, k0, lane);
            sc[j] = __builtin_amdgcn_wmma_f32_16x16x32_bf16(false, aq, false, bk, (short)0, sc[j], false, false);
        }
    }
    const int nlo = lane & 15;
    const int mhi = (lane >> 4) << 3;
#pragma unroll
    for (int j = 0; j < 4; ++j) {
        int n = j * 16 + nlo;
#pragma unroll
        for (int v = 0; v < 8; ++v)
            scS[(m0 + mhi + v) * 68 + n] = sc[j][v] * 0.125f;   // 1/sqrt(64)
    }
    __syncthreads();

    // per-row softmax (one row per thread, rows 0..63)
    if (tid < 64) {
        float mx = -1e30f;
        for (int c = 0; c < 64; ++c) mx = fmaxf(mx, scS[tid * 68 + c]);
        float sum = 0.0f;
        for (int c = 0; c < 64; ++c) {
            float ev = __expf(scS[tid * 68 + c] - mx);
            scS[tid * 68 + c] = ev;
            sum += ev;
        }
        float inv = 1.0f / sum;
        for (int c = 0; c < 64; ++c)
            Pm[tid * 64 + c] = (__bf16)(scS[tid * 68 + c] * inv);
    }
    __syncthreads();

    // out = P @ V   (A = Pm rows b x K=c ; B = Vt rows hd x K=c)
    v8f o[4] = {{}, {}, {}, {}};
#pragma unroll
    for (int c0 = 0; c0 < 64; c0 += 32) {
        v16bf ap;
        {
            int r  = m0 + (lane & 15);
            int kb = c0 + ((lane >> 4) << 3);
#pragma unroll
            for (int i = 0; i < 8; ++i) {
                ap[i]     = Pm[r * 64 + kb + i];
                ap[i + 8] = Pm[r * 64 + kb + 16 + i];
            }
        }
#pragma unroll
        for (int j = 0; j < 4; ++j) {
            v16bf bv;
            int rr = j * 16 + (lane & 15);           // hd
            int kb = c0 + ((lane >> 4) << 3);
#pragma unroll
            for (int i = 0; i < 8; ++i) {
                bv[i]     = Vt[rr * 64 + kb + i];
                bv[i + 8] = Vt[rr * 64 + kb + 16 + i];
            }
            o[j] = __builtin_amdgcn_wmma_f32_16x16x32_bf16(false, ap, false, bv, (short)0, o[j], false, false);
        }
    }
#pragma unroll
    for (int j = 0; j < 4; ++j) {
        int n = j * 16 + nlo;
#pragma unroll
        for (int v = 0; v < 8; ++v)
            Ob[base + (size_t)(m0 + mhi + v) * D + n] = (__bf16)o[j][v];
    }
}

// ---------------------------------------------------------------------------
// x = LayerNorm(x + y) * g + b ; writes f32 and bf16 copies. One block per row.
// ---------------------------------------------------------------------------
__global__ __launch_bounds__(256)
void add_ln_kernel(const float* __restrict__ xin, const float* __restrict__ yin,
                   const float* __restrict__ g, const float* __restrict__ be,
                   float* __restrict__ xout, __bf16* __restrict__ xbout)
{
    const int row = blockIdx.x;
    const int tid = threadIdx.x;
    __shared__ float red[256];
    float vals[3];
    float s = 0.0f;
#pragma unroll
    for (int i = 0; i < 3; ++i) {
        int d = tid + i * 256;
        float v = xin[(size_t)row * D + d] + yin[(size_t)row * D + d];
        vals[i] = v; s += v;
    }
    red[tid] = s; __syncthreads();
    for (int o = 128; o > 0; o >>= 1) { if (tid < o) red[tid] += red[tid + o]; __syncthreads(); }
    float mean = red[0] / D;
    __syncthreads();
    float s2 = 0.0f;
#pragma unroll
    for (int i = 0; i < 3; ++i) { float d0 = vals[i] - mean; s2 += d0 * d0; }
    red[tid] = s2; __syncthreads();
    for (int o = 128; o > 0; o >>= 1) { if (tid < o) red[tid] += red[tid + o]; __syncthreads(); }
    float inv = rsqrtf(red[0] / D + 1e-5f);
#pragma unroll
    for (int i = 0; i < 3; ++i) {
        int d = tid + i * 256;
        float v = (vals[i] - mean) * inv * g[d] + be[d];
        xout[(size_t)row * D + d]  = v;
        xbout[(size_t)row * D + d] = (__bf16)v;
    }
}

// ---------------------------------------------------------------------------
// Mean over the sequence axis: xm[b,d] = (1/S) sum_s x[s,b,d]
// ---------------------------------------------------------------------------
__global__ __launch_bounds__(256)
void mean_s_kernel(const float* __restrict__ xf, float* __restrict__ xm)
{
    int i = blockIdx.x * blockDim.x + threadIdx.x;     // b*D + d
    if (i >= Bb * D) return;
    float s = 0.0f;
    for (int si = 0; si < S; ++si) s += xf[(size_t)si * Bb * D + i];
    xm[i] = s * (1.0f / S);
}

// ---------------------------------------------------------------------------
// Classifier head: out[b,c] = xm[b,:] . Wfc[c,:] + bfc[c]  (Wfc is LoRA-folded)
// ---------------------------------------------------------------------------
__global__ __launch_bounds__(256)
void fc_kernel(const float* __restrict__ xm, const float* __restrict__ Wfc,
               const float* __restrict__ bfc, float* __restrict__ out)
{
    const int bc = blockIdx.x;            // b*C + c
    const int b = bc / C, c = bc % C;
    const int tid = threadIdx.x;
    __shared__ float red[256];
    float s = 0.0f;
    for (int d = tid; d < D; d += 256)
        s += xm[(size_t)b * D + d] * Wfc[(size_t)c * D + d];
    red[tid] = s; __syncthreads();
    for (int o = 128; o > 0; o >>= 1) { if (tid < o) red[tid] += red[tid + o]; __syncthreads(); }
    if (tid == 0) out[bc] = red[0] + bfc[c];
}

// ---------------------------------------------------------------------------
// Driver
// ---------------------------------------------------------------------------
extern "C" void kernel_launch(void* const* d_in, const int* in_sizes, int n_in,
                              void* d_out, int out_size, void* d_ws, size_t ws_size,
                              hipStream_t stream)
{
    const int*   tokens = (const int*)  d_in[0];
    const float* embW   = (const float*)d_in[1];
    const float* embA   = (const float*)d_in[2];
    const float* embB   = (const float*)d_in[3];
    const float* Wq = (const float*)d_in[4],  *bq = (const float*)d_in[5],
               * Aq = (const float*)d_in[6],  *Bq = (const float*)d_in[7];
    const float* Wk = (const float*)d_in[8],  *bk = (const float*)d_in[9],
               * Ak = (const float*)d_in[10], *Bk = (const float*)d_in[11];
    const float* Wv = (const float*)d_in[12], *bv = (const float*)d_in[13],
               * Av = (const float*)d_in[14], *Bv = (const float*)d_in[15];
    const float* Wo = (const float*)d_in[16], *bo = (const float*)d_in[17],
               * Ao = (const float*)d_in[18], *Bo = (const float*)d_in[19];
    const float* W1 = (const float*)d_in[20], *b1 = (const float*)d_in[21],
               * A1 = (const float*)d_in[22], *B1 = (const float*)d_in[23];
    const float* W2 = (const float*)d_in[24], *b2 = (const float*)d_in[25],
               * A2 = (const float*)d_in[26], *B2 = (const float*)d_in[27];
    const float* g1 = (const float*)d_in[28], *be1 = (const float*)d_in[29];
    const float* g2 = (const float*)d_in[30], *be2 = (const float*)d_in[31];
    const float* fcW = (const float*)d_in[32], *fcb = (const float*)d_in[33],
               * fcA = (const float*)d_in[34], *fcB = (const float*)d_in[35];

    // workspace carve-out
    char* ws = (char*)d_ws;
    size_t off = 0;
    auto alloc = [&](size_t bytes) -> char* {
        char* p = ws + off;
        off = (off + bytes + 255) & ~(size_t)255;
        return p;
    };
    float*  xf   = (float*) alloc((size_t)M * D * 4);
    __bf16* xb   = (__bf16*)alloc((size_t)M * D * 2);
    __bf16* qb   = (__bf16*)alloc((size_t)M * D * 2);
    __bf16* kb2  = (__bf16*)alloc((size_t)M * D * 2);
    __bf16* vb2  = (__bf16*)alloc((size_t)M * D * 2);
    __bf16* aob  = (__bf16*)alloc((size_t)M * D * 2);
    __bf16* h1b  = (__bf16*)alloc((size_t)M * HID * 2);
    float*  yf   = (float*) alloc((size_t)M * D * 4);
    __bf16* wqE  = (__bf16*)alloc((size_t)L * D * D * 2);
    __bf16* wkE  = (__bf16*)alloc((size_t)L * D * D * 2);
    __bf16* wvE  = (__bf16*)alloc((size_t)L * D * D * 2);
    __bf16* woE  = (__bf16*)alloc((size_t)L * D * D * 2);
    __bf16* w1E  = (__bf16*)alloc((size_t)L * HID * D * 2);
    __bf16* w2E  = (__bf16*)alloc((size_t)L * D * HID * 2);
    float*  fcE  = (float*) alloc((size_t)C * D * 4);
    float*  xm   = (float*) alloc((size_t)Bb * D * 4);

    auto cdiv = [](long long a, long long b) -> int { return (int)((a + b - 1) / b); };

    // 1) fold LoRA into weights
    for (int l = 0; l < L; ++l) {
        fold_lora_bf16<<<cdiv((long long)D * D, 256), 256, 0, stream>>>(
            Wq + (size_t)l * D * D, Aq + (size_t)l * R * D, Bq + (size_t)l * D * R,
            wqE + (size_t)l * D * D, D, D);
        fold_lora_bf16<<<cdiv((long long)D * D, 256), 256, 0, stream>>>(
            Wk + (size_t)l * D * D, Ak + (size_t)l * R * D, Bk + (size_t)l * D * R,
            wkE + (size_t)l * D * D, D, D);
        fold_lora_bf16<<<cdiv((long long)D * D, 256), 256, 0, stream>>>(
            Wv + (size_t)l * D * D, Av + (size_t)l * R * D, Bv + (size_t)l * D * R,
            wvE + (size_t)l * D * D, D, D);
        fold_lora_bf16<<<cdiv((long long)D * D, 256), 256, 0, stream>>>(
            Wo + (size_t)l * D * D, Ao + (size_t)l * R * D, Bo + (size_t)l * D * R,
            woE + (size_t)l * D * D, D, D);
        fold_lora_bf16<<<cdiv((long long)HID * D, 256), 256, 0, stream>>>(
            W1 + (size_t)l * HID * D, A1 + (size_t)l * R * D, B1 + (size_t)l * HID * R,
            w1E + (size_t)l * HID * D, HID, D);
        fold_lora_bf16<<<cdiv((long long)D * HID, 256), 256, 0, stream>>>(
            W2 + (size_t)l * D * HID, A2 + (size_t)l * R * HID, B2 + (size_t)l * D * R,
            w2E + (size_t)l * D * HID, D, HID);
    }
    fold_lora_f32<<<cdiv((long long)C * D, 256), 256, 0, stream>>>(fcW, fcA, fcB, fcE, C, D);

    // 2) LoRA embedding gather -> x (S,B,D) f32 + bf16
    embed_kernel<<<M, 256, 0, stream>>>(tokens, embW, embA, embB, xf, xb);

    const dim3 gD(D / 128, M / 64);     // (6, 512)
    const dim3 gH(HID / 128, M / 64);   // (24, 512)

    // 3) transformer layers
    for (int l = 0; l < L; ++l) {
        gemm_bf16_wmma<<<gD, 256, 0, stream>>>(xb, wqE + (size_t)l * D * D, bq + (size_t)l * D,
                                               nullptr, qb, M, D, D, 0);
        gemm_bf16_wmma<<<gD, 256, 0, stream>>>(xb, wkE + (size_t)l * D * D, bk + (size_t)l * D,
                                               nullptr, kb2, M, D, D, 0);
        gemm_bf16_wmma<<<gD, 256, 0, stream>>>(xb, wvE + (size_t)l * D * D, bv + (size_t)l * D,
                                               nullptr, vb2, M, D, D, 0);
        attention_wmma<<<S * H, 128, 0, stream>>>(qb, kb2, vb2, aob);
        gemm_bf16_wmma<<<gD, 256, 0, stream>>>(aob, woE + (size_t)l * D * D, bo + (size_t)l * D,
                                               yf, nullptr, M, D, D, 0);
        add_ln_kernel<<<M, 256, 0, stream>>>(xf, yf, g1 + (size_t)l * D, be1 + (size_t)l * D, xf, xb);
        gemm_bf16_wmma<<<gH, 256, 0, stream>>>(xb, w1E + (size_t)l * HID * D, b1 + (size_t)l * HID,
                                               nullptr, h1b, M, HID, D, 1);
        gemm_bf16_wmma<<<gD, 256, 0, stream>>>(h1b, w2E + (size_t)l * D * HID, b2 + (size_t)l * D,
                                               yf, nullptr, M, D, HID, 0);
        add_ln_kernel<<<M, 256, 0, stream>>>(xf, yf, g2 + (size_t)l * D, be2 + (size_t)l * D, xf, xb);
    }

    // 4) head
    mean_s_kernel<<<cdiv((long long)Bb * D, 256), 256, 0, stream>>>(xf, xm);
    fc_kernel<<<Bb * C, 256, 0, stream>>>(xm, fcE, fcb, (float*)d_out);
}
